// CANDY_41077067219071
// MI455X (gfx1250) — matrix operand order repaired and checked by reading, not compiled, not run on hardware
//
#include <hip/hip_runtime.h>

typedef __attribute__((ext_vector_type(16))) __bf16 v16bf;
typedef __attribute__((ext_vector_type(8)))  float  v8f;
typedef __attribute__((__vector_size__(16))) int    v4i;

#define H_DIM 1024
#define I_DIM 1024
#define C_DIM 64

#define BM 128
#define BN 256
#define BK 32

#if __has_builtin(__builtin_amdgcn_global_load_async_to_lds_b128)
#define HAS_ASYNC 1
#else
#define HAS_ASYNC 0
#endif

typedef __attribute__((address_space(1))) v4i* gas_v4i;   // global (prints as "__device__")
typedef __attribute__((address_space(3))) v4i* las_v4i;   // LDS    (prints as "__shared__")

__device__ __forceinline__ gas_v4i to_gas(const void* p) {
    return (gas_v4i)(uintptr_t)p;              // generic -> global (64-bit)
}
__device__ __forceinline__ las_v4i to_las(const void* p) {
    return (las_v4i)(unsigned)(uintptr_t)p;    // generic LDS addr low 32 bits = LDS offset
}

#if HAS_ASYNC
#if __has_builtin(__builtin_amdgcn_s_wait_asynccnt)
#define WAIT_ASYNC() __builtin_amdgcn_s_wait_asynccnt(0)
#else
#define WAIT_ASYNC() asm volatile("s_wait_asynccnt 0x0" ::: "memory")
#endif
#else
#define WAIT_ASYNC()
#endif

__device__ __forceinline__ float actf(float x) {
    // (|x+1|-|x-1|)/2 == clamp(x,-1,1)
    return fminf(fmaxf(x, -1.0f), 1.0f);
}

union FragU { uint4 q[2]; v16bf v; };
union PackU { __bf16 h[8]; uint4 q; };

// ---------------- prologue kernels ----------------

// Wp_eff = tril(Wp) with diag=clamp(diag,0,1)+Wp_diag, converted to bf16
__global__ void k_prep_wp(const float* __restrict__ Wp, const float* __restrict__ Wdiag,
                          __bf16* __restrict__ out) {
    int idx = blockIdx.x * 256 + threadIdx.x;
    int h = idx >> 10, k = idx & 1023;
    float v;
    if (k < h)       v = Wp[idx];
    else if (k == h) v = fminf(fmaxf(Wp[idx], 0.0f), 1.0f) + Wdiag[h];
    else             v = 0.0f;
    out[idx] = (__bf16)v;
}

__global__ void k_cvt(const float* __restrict__ in, __bf16* __restrict__ out, int n) {
    int idx = blockIdx.x * 256 + threadIdx.x;
    if (idx < n) out[idx] = (__bf16)in[idx];
}

// psetT[c][i][h] = bf16( x[c][h][i] * act(p_mask[h][i]) )  (LDS-tiled transpose)
__global__ void k_psetT(const float* __restrict__ x, const float* __restrict__ pmask,
                        __bf16* __restrict__ out) {
    __shared__ float t[32][33];
    const int i0 = blockIdx.x * 32, h0 = blockIdx.y * 32, c = blockIdx.z;
    const int tx = threadIdx.x, ty = threadIdx.y;
    const float* xc = x + (size_t)c * H_DIM * I_DIM;
#pragma unroll
    for (int r = 0; r < 4; ++r) {
        int h = h0 + ty + r * 8;
        size_t off = (size_t)h * I_DIM + i0 + tx;
        t[ty + r * 8][tx] = xc[off] * actf(pmask[off]);
    }
    __syncthreads();
    __bf16* oc = out + (size_t)c * H_DIM * I_DIM;
#pragma unroll
    for (int r = 0; r < 4; ++r) {
        int i = i0 + ty + r * 8;
        oc[(size_t)i * H_DIM + h0 + tx] = (__bf16)t[tx][ty + r * 8];
    }
}

// ---------------- core NT-GEMM with WMMA ----------------
// R[q][p] = sum_k P[p][k] * Q[q][k]; P per-channel (1024x1024 bf16), Q shared weight.
// MODE 1: act -> bf16 transposed store (outT[(c,q,p)])
// MODE 2: +bias, act -> fp32 natural store (outF[(c,p,q)]) AND bf16 transposed store
// MODE 4: +bias, act, + resid[(c,p,q)] -> fp32 natural store
template <int MODE>
__global__ __launch_bounds__(256)
void gemm_nt(const __bf16* __restrict__ P, const __bf16* __restrict__ Q,
             const float* __restrict__ bias,
             float* __restrict__ outF, __bf16* __restrict__ outT,
             const float* __restrict__ resid) {
    constexpr int M = 1024, N = 1024, K = 1024;
    __shared__ __bf16 lsP[2][BM][BK + 8];
    __shared__ __bf16 lsQ[2][BN][BK + 8];

    const int tid = threadIdx.x;
    const int lane = tid & 31;
    const int wid = tid >> 5;
    const int wp = wid & 1;        // wave p-slot (0..1)
    const int wq = wid >> 1;       // wave q-slot (0..3)
    const int half = lane >> 4;    // hi/lo lane group
    const int lm = lane & 15;
    const int c = blockIdx.z;
    const int p0 = blockIdx.y * BM;
    const int q0 = blockIdx.x * BN;

    const __bf16* Pc = P + (size_t)c * M * K;

    v8f acc[4][4];
#pragma unroll
    for (int i = 0; i < 4; ++i)
#pragma unroll
        for (int j = 0; j < 4; ++j) acc[i][j] = (v8f)0.0f;

    const int lr = tid >> 2;          // 0..63
    const int ls = (tid & 3) * 8;     // bf16 element offset 0/8/16/24

    // stage a 128x32 P-tile + 256x32 Q-tile into LDS buffer b (6 x 16B per thread)
    auto load_tile = [&](int b, int k0) {
#if HAS_ASYNC
        __builtin_amdgcn_global_load_async_to_lds_b128(
            to_gas(Pc + (size_t)(p0 + lr) * K + k0 + ls), to_las(&lsP[b][lr][ls]), 0, 0);
        __builtin_amdgcn_global_load_async_to_lds_b128(
            to_gas(Pc + (size_t)(p0 + lr + 64) * K + k0 + ls), to_las(&lsP[b][lr + 64][ls]), 0, 0);
#pragma unroll
        for (int i = 0; i < 4; ++i)
            __builtin_amdgcn_global_load_async_to_lds_b128(
                to_gas(Q + (size_t)(q0 + lr + 64 * i) * K + k0 + ls),
                to_las(&lsQ[b][lr + 64 * i][ls]), 0, 0);
#else
        *(uint4*)&lsP[b][lr][ls]      = *(const uint4*)(Pc + (size_t)(p0 + lr) * K + k0 + ls);
        *(uint4*)&lsP[b][lr + 64][ls] = *(const uint4*)(Pc + (size_t)(p0 + lr + 64) * K + k0 + ls);
#pragma unroll
        for (int i = 0; i < 4; ++i)
            *(uint4*)&lsQ[b][lr + 64 * i][ls] =
                *(const uint4*)(Q + (size_t)(q0 + lr + 64 * i) * K + k0 + ls);
#endif
    };

    constexpr int nK = K / BK;
    load_tile(0, 0);

    for (int kt = 0; kt < nK; ++kt) {
        WAIT_ASYNC();
        __syncthreads();           // buffer kt&1 ready on all waves; kt+1 buffer free
        if (kt + 1 < nK) load_tile((kt + 1) & 1, (kt + 1) * BK);

        const int b = kt & 1;
        // build fragments (ISA 16-bit A 16x32 / B 32x16 lane layouts)
        v16bf a[4], bfr[4];
#pragma unroll
        for (int i = 0; i < 4; ++i) {
            const __bf16* rp = &lsP[b][wp * 64 + i * 16 + lm][0];
            FragU u;
            u.q[0] = *(const uint4*)(rp + (half ? 8 : 0));    // K 0-7 | 8-15
            u.q[1] = *(const uint4*)(rp + (half ? 24 : 16));  // K 16-23 | 24-31
            a[i] = u.v;
        }
#pragma unroll
        for (int j = 0; j < 4; ++j) {
            const __bf16* rp = &lsQ[b][wq * 64 + j * 16 + lm][0];
            FragU u;
            u.q[0] = *(const uint4*)(rp + (half ? 8 : 0));    // K 0-7 | 8-15
            u.q[1] = *(const uint4*)(rp + (half ? 24 : 16));  // K 16-23 | 24-31
            bfr[j] = u.v;
        }
#pragma unroll
        for (int i = 0; i < 4; ++i)
#pragma unroll
            for (int j = 0; j < 4; ++j)
                acc[i][j] = __builtin_amdgcn_wmma_f32_16x16x32_bf16(
                    false, a[i], false, bfr[j], (short)0, acc[i][j], false, false);
    }

    // epilogue
    const size_t cMN = (size_t)c * M * N;
#pragma unroll
    for (int i = 0; i < 4; ++i) {
        const int pBase = p0 + wp * 64 + i * 16 + half * 8;  // 8 consecutive p per lane
#pragma unroll
        for (int j = 0; j < 4; ++j) {
            const int q = q0 + wq * 64 + j * 16 + lm;
            float bv = 0.0f;
            if constexpr (MODE == 2 || MODE == 4) bv = bias[q];
            float v[8];
#pragma unroll
            for (int r = 0; r < 8; ++r) v[r] = actf(acc[i][j][r] + bv);

            if constexpr (MODE == 1 || MODE == 2) {
                PackU pk;
#pragma unroll
                for (int r = 0; r < 8; ++r) pk.h[r] = (__bf16)v[r];
                *(uint4*)(outT + cMN + (size_t)q * M + pBase) = pk.q;
            }
            if constexpr (MODE == 2) {
#pragma unroll
                for (int r = 0; r < 8; ++r)
                    outF[cMN + (size_t)(pBase + r) * N + q] = v[r];
            }
            if constexpr (MODE == 4) {
#pragma unroll
                for (int r = 0; r < 8; ++r) {
                    size_t idx = cMN + (size_t)(pBase + r) * N + q;
                    outF[idx] = v[r] + resid[idx];
                }
            }
        }
    }
}

// ---------------- host launch ----------------
extern "C" void kernel_launch(void* const* d_in, const int* in_sizes, int n_in,
                              void* d_out, int out_size, void* d_ws, size_t ws_size,
                              hipStream_t stream) {
    const float* x      = (const float*)d_in[0];
    const float* pmask  = (const float*)d_in[1];
    const float* Wp     = (const float*)d_in[2];
    const float* Wdiag  = (const float*)d_in[3];
    const float* Wzp    = (const float*)d_in[4];
    const float* plw    = (const float*)d_in[5];
    const float* plb    = (const float*)d_in[6];
    const float* zlw    = (const float*)d_in[7];
    const float* zlb    = (const float*)d_in[8];
    float* out = (float*)d_out;

    const size_t MN  = (size_t)H_DIM * I_DIM;          // 1M elems
    const size_t CMN = (size_t)C_DIM * MN;             // 64M elems
    char* ws = (char*)d_ws;
    __bf16* wEff = (__bf16*)(ws);
    __bf16* wzpB = (__bf16*)(ws + 2 * MN);             // 2 bytes/elem
    __bf16* plwB = (__bf16*)(ws + 4 * MN);
    __bf16* zlwB = (__bf16*)(ws + 6 * MN);
    __bf16* bufA = (__bf16*)(ws + 8 * MN);             // C*H*I bf16 (128MB)
    __bf16* bufB = (__bf16*)(ws + 8 * MN + 2 * CMN);   // C*H*I bf16 (128MB)

    const int n = (int)MN;
    k_prep_wp<<<n / 256, 256, 0, stream>>>(Wp, Wdiag, wEff);
    k_cvt<<<n / 256, 256, 0, stream>>>(Wzp, wzpB, n);
    k_cvt<<<n / 256, 256, 0, stream>>>(plw, plwB, n);
    k_cvt<<<n / 256, 256, 0, stream>>>(zlw, zlwB, n);

    k_psetT<<<dim3(I_DIM / 32, H_DIM / 32, C_DIM), dim3(32, 8), 0, stream>>>(x, pmask, bufA);

    dim3 grid(I_DIM / BN, H_DIM / BM, C_DIM), block(256);
    // stage1: act(t)(h,i) = psetT (x) Wp_eff
    gemm_nt<1><<<grid, block, 0, stream>>>(bufA, wEff, nullptr, nullptr, bufB, nullptr);
    // stage2: p_out fp32 (h,j) -> d_out; bf16 p_outT (j,h) -> bufA
    gemm_nt<2><<<grid, block, 0, stream>>>(bufB, plwB, plb, out, bufA, nullptr);
    // stage3: act(z)(h,i) = p_outT (x) Wzp
    gemm_nt<1><<<grid, block, 0, stream>>>(bufA, wzpB, nullptr, nullptr, bufB, nullptr);
    // stage4: out = p_out + act(act(z) @ zlw^T + b)
    gemm_nt<4><<<grid, block, 0, stream>>>(bufB, zlwB, zlb, out, nullptr, out);
}
